// DeepseekSparseAttention_64175401337410
// MI455X (gfx1250) — matrix operand order repaired and checked by reading, not compiled, not run on hardware
//
#include <hip/hip_runtime.h>
#include <hip/hip_bf16.h>
#include <math.h>

// ---------------------------------------------------------------------------
// Types for CDNA5 WMMA (wave32, 16x16x32 bf16 -> f32)
// ---------------------------------------------------------------------------
typedef __bf16 bf16_t;
typedef __attribute__((ext_vector_type(16))) __bf16 v16bf;
typedef __attribute__((ext_vector_type(8)))  __bf16 v8bf;
typedef __attribute__((ext_vector_type(8)))  float  v8f;

// gcc-style vector matching clang's builtin "V4i" canonical type
typedef int gv4i __attribute__((vector_size(16)));
typedef __attribute__((address_space(1))) gv4i* as1_v4i_ptr;
typedef __attribute__((address_space(3))) gv4i* as3_v4i_ptr;

#define NBATCH 8
#define SEQ    8192
#define DMODEL 2048
#define DLAT   512
#define NHEADS 16
#define DHEAD  128
#define TOPK   2048
#define KVN    1024              // 2*DLAT (K_down | V_down concat)
#define MROWS  (NBATCH * SEQ)    // 65536

#if __has_builtin(__builtin_amdgcn_global_load_async_to_lds_b128)
#define HAVE_ASYNC_LDS 1
#else
#define HAVE_ASYNC_LDS 0
#endif

__device__ inline void wait_async0() {
#if __has_builtin(__builtin_amdgcn_s_wait_asynccnt)
  __builtin_amdgcn_s_wait_asynccnt(0);
#else
  asm volatile("s_wait_asynccnt 0" ::: "memory");
#endif
}

__device__ inline v8f vzero8() {
  v8f z = {0.f, 0.f, 0.f, 0.f, 0.f, 0.f, 0.f, 0.f};
  return z;
}

__device__ inline v16bf cat16(v8bf lo, v8bf hi) {
  return __builtin_shufflevector(lo, hi, 0, 1, 2, 3, 4, 5, 6, 7,
                                         8, 9, 10, 11, 12, 13, 14, 15);
}

__device__ inline v8f wmma_bf16f32(v16bf a, v16bf b, v8f c) {
  // (neg_a, A, neg_b, B, c_mod, C, reuse_a, reuse_b)
  return __builtin_amdgcn_wmma_f32_16x16x32_bf16(false, a, false, b,
                                                 (short)0, c, false, false);
}

__device__ inline v16bf cvt16(float4 f0, float4 f1, float4 f2, float4 f3) {
  v16bf a;
  a[0]  = (bf16_t)f0.x; a[1]  = (bf16_t)f0.y; a[2]  = (bf16_t)f0.z; a[3]  = (bf16_t)f0.w;
  a[4]  = (bf16_t)f1.x; a[5]  = (bf16_t)f1.y; a[6]  = (bf16_t)f1.z; a[7]  = (bf16_t)f1.w;
  a[8]  = (bf16_t)f2.x; a[9]  = (bf16_t)f2.y; a[10] = (bf16_t)f2.z; a[11] = (bf16_t)f2.w;
  a[12] = (bf16_t)f3.x; a[13] = (bf16_t)f3.y; a[14] = (bf16_t)f3.z; a[15] = (bf16_t)f3.w;
  return a;
}

// Emulated fp8 e4m3fn round-trip (RNE; saturate at 448; denorm step 2^-9).
__device__ inline float q8(float x) {
  unsigned u = __float_as_uint(x);
  unsigned s = u & 0x80000000u;
  unsigned a = u & 0x7fffffffu;
  float ax = __uint_as_float(a);
  float r;
  if (ax < 0.015625f) {                       // below 2^-6: fp8 denormal range
    r = rintf(ax * 512.0f) * (1.0f / 512.0f);
  } else {
    if (a > 0x43e00000u) a = 0x43e00000u;     // clamp to 448
    unsigned lsb  = 1u << 20;                 // keep 3 mantissa bits
    unsigned bias = (lsb >> 1) - 1u + ((a >> 20) & 1u);
    a = (a + bias) & ~(lsb - 1u);
    if (a > 0x43e00000u) a = 0x43e00000u;
    r = __uint_as_float(a);
  }
  return __uint_as_float(__float_as_uint(r) | s);
}

// ---------------------------------------------------------------------------
// 0) One-shot: Wt[n][k] = bf16(Wkv_down[k][n])  (1024 x 2048 bf16, 4 MB)
//    Puts B in the exact per-lane fragment order so the GEMM needs no LDS.
// ---------------------------------------------------------------------------
__global__ __launch_bounds__(256) void k_wprep(const float* __restrict__ W,
                                               bf16_t* __restrict__ Wt) {
  const int n  = blockIdx.x * 64 + (threadIdx.x & 63);
  const int k0 = (threadIdx.x >> 6) * 512;
  bf16_t* dst = Wt + (size_t)n * DMODEL;
  for (int k = k0; k < k0 + 512; ++k)
    dst[k] = (bf16_t)W[(size_t)k * KVN + n];
}

// ---------------------------------------------------------------------------
// 1) KV = x @ Wkv_down + bkv_down   (M=65536, K=2048, N=1024) -> bf16
//    LDS-free, barrier-free: 128x128 tile/block, wave owns a 16-row stripe;
//    A fragments converted in-register, B fragments are contiguous loads
//    from the pre-transposed bf16 weights. 8 WMMAs per K-step per wave.
// ---------------------------------------------------------------------------
__global__ __launch_bounds__(256) void k_kvdown(const float* __restrict__ x,
                                                const bf16_t* __restrict__ Wt,
                                                const float* __restrict__ bias,
                                                bf16_t* __restrict__ kv) {
  const int nb = blockIdx.x;                      // 0..7
  const int mb = blockIdx.y;                      // 0..511
  const int wave = threadIdx.x >> 5;
  const int lane = threadIdx.x & 31;
  const int nloc = lane & 15;
  const int khA  = (lane < 16) ? 0 : 8;
  const int khB  = (lane < 16) ? 0 : 8;
  const int m0 = mb * 128;
  const int n0 = nb * 128;

  const float*  xr = x  + (size_t)(m0 + wave * 16 + nloc) * DMODEL;
  const bf16_t* wr = Wt + (size_t)(n0 + nloc) * DMODEL;

  v8f acc[8];
#pragma unroll
  for (int i = 0; i < 8; ++i) acc[i] = vzero8();

  for (int k0 = 0; k0 < DMODEL; k0 += 32) {
    const float4 f0 = *(const float4*)(xr + k0 + khA);
    const float4 f1 = *(const float4*)(xr + k0 + khA + 4);
    const float4 f2 = *(const float4*)(xr + k0 + khA + 16);
    const float4 f3 = *(const float4*)(xr + k0 + khA + 20);
    if (k0 + 32 < DMODEL)
      __builtin_prefetch(xr + k0 + 32 + khA, 0, 1);
    const v16bf afrag = cvt16(f0, f1, f2, f3);
#pragma unroll
    for (int nt = 0; nt < 8; ++nt) {
      const bf16_t* bp = wr + (size_t)nt * 16 * DMODEL + k0 + khB;
      v16bf bfrag = cat16(*(const v8bf*)bp, *(const v8bf*)(bp + 16));
      acc[nt] = wmma_bf16f32(afrag, bfrag, acc[nt]);
    }
  }

#pragma unroll
  for (int nt = 0; nt < 8; ++nt) {
#pragma unroll
    for (int r = 0; r < 8; ++r) {
      int gm = m0 + wave * 16 + r + ((lane < 16) ? 0 : 8);
      int gn = n0 + nt * 16 + nloc;
      kv[(size_t)gm * KVN + gn] = (bf16_t)(acc[nt][r] + bias[gn]);
    }
  }
}

// ---------------------------------------------------------------------------
// 2) q_idx[b,l] = q8(x[b,-1,:]) . q8(Wq_down[:,l]) + q8(bq_down[l])
// ---------------------------------------------------------------------------
__global__ __launch_bounds__(256) void k_qidx(const float* __restrict__ x,
                                              const float* __restrict__ Wq_down,
                                              const float* __restrict__ bq_down,
                                              float* __restrict__ q_idx) {
  const int b = blockIdx.x, tid = threadIdx.x;
  __shared__ float xs[DMODEL];
  const float* xr = x + ((size_t)b * SEQ + (SEQ - 1)) * DMODEL;
  for (int d = tid; d < DMODEL; d += 256) xs[d] = q8(xr[d]);
  __syncthreads();
  for (int l = tid; l < DLAT; l += 256) {
    float accv = q8(bq_down[l]);
    for (int d = 0; d < DMODEL; ++d)
      accv += xs[d] * q8(Wq_down[(size_t)d * DLAT + l]);
    q_idx[b * DLAT + l] = accv;
  }
}

// ---------------------------------------------------------------------------
// 3) scores[b,s] = relu( q_idx[b,:] . q8(K_down[b,s,:]) )
// ---------------------------------------------------------------------------
__global__ __launch_bounds__(256) void k_scores(const bf16_t* __restrict__ kv,
                                                const float* __restrict__ q_idx,
                                                float* __restrict__ scores) {
  const int b = blockIdx.y;
  const int s = blockIdx.x * 256 + threadIdx.x;
  __shared__ float qs[DLAT];
  for (int l = threadIdx.x; l < DLAT; l += 256) qs[l] = q_idx[b * DLAT + l];
  __syncthreads();
  const bf16_t* kr = kv + (size_t)(b * SEQ + s) * KVN;   // K_down = cols 0..511
  float accv = 0.f;
  for (int l0 = 0; l0 < DLAT; l0 += 8) {
    const v8bf kb = *(const v8bf*)(kr + l0);
#pragma unroll
    for (int j = 0; j < 8; ++j) accv += qs[l0 + j] * q8((float)kb[j]);
  }
  scores[b * SEQ + s] = accv > 0.f ? accv : 0.f;
}

// ---------------------------------------------------------------------------
// 4) top-k (k=2048 of 8192) per batch via 32-bit radix threshold select.
// ---------------------------------------------------------------------------
__global__ __launch_bounds__(1024) void k_topk(const float* __restrict__ scores,
                                               int* __restrict__ idx) {
  __shared__ unsigned sv[SEQ];
  __shared__ int cnt, pos1, pos2;
  const int b = blockIdx.x, tid = threadIdx.x;
  for (int i = tid; i < SEQ; i += 1024)
    sv[i] = __float_as_uint(scores[b * SEQ + i]);
  __syncthreads();
  unsigned thr = 0u;
  for (int bit = 31; bit >= 0; --bit) {
    unsigned cand = thr | (1u << bit);
    if (tid == 0) cnt = 0;
    __syncthreads();
    int local = 0;
    for (int i = tid; i < SEQ; i += 1024) local += (sv[i] >= cand) ? 1 : 0;
    atomicAdd(&cnt, local);
    __syncthreads();
    if (cnt >= TOPK) thr = cand;
    __syncthreads();
  }
  if (tid == 0) pos1 = 0;
  __syncthreads();
  for (int i = tid; i < SEQ; i += 1024) {
    if (sv[i] > thr) {
      int p = atomicAdd(&pos1, 1);
      if (p < TOPK) idx[b * TOPK + p] = i;
    }
  }
  __syncthreads();
  if (tid == 0) pos2 = pos1;
  __syncthreads();
  for (int i = tid; i < SEQ; i += 1024) {
    if (sv[i] == thr) {
      int p = atomicAdd(&pos2, 1);
      if (p < TOPK) idx[b * TOPK + p] = i;
    }
  }
}

// ---------------------------------------------------------------------------
// 5) q[b,:] = x[b,-1,:] @ Wq + bq
// ---------------------------------------------------------------------------
__global__ __launch_bounds__(256) void k_qproj(const float* __restrict__ x,
                                               const float* __restrict__ Wq,
                                               const float* __restrict__ bq,
                                               float* __restrict__ q) {
  const int b = blockIdx.y;
  const int j = blockIdx.x * 256 + threadIdx.x;
  __shared__ float xs[DMODEL];
  const float* xr = x + ((size_t)b * SEQ + (SEQ - 1)) * DMODEL;
  for (int d = threadIdx.x; d < DMODEL; d += 256) xs[d] = xr[d];
  __syncthreads();
  float accv = bq[j];
  for (int d = 0; d < DMODEL; ++d) accv += xs[d] * Wq[(size_t)d * DMODEL + j];
  q[b * DMODEL + j] = accv;
}

// ---------------------------------------------------------------------------
// 6) Absorption: q_abs[b,h,l] = sum_d q[b,h*128+d] * Wk_up[l, h*128+d]
//    qb[b,h] = q[b,h,:] . bk_up[h,:]     (Wk_up = Wkv_up[:, :2048])
// ---------------------------------------------------------------------------
__global__ __launch_bounds__(256) void k_qabs(const float* __restrict__ q,
                                              const float* __restrict__ Wkv_up,
                                              const float* __restrict__ bkv_up,
                                              bf16_t* __restrict__ q_abs,
                                              float* __restrict__ qb) {
  const int b = blockIdx.x, tid = threadIdx.x;
  __shared__ float qs[DMODEL];
  for (int d = tid; d < DMODEL; d += 256) qs[d] = q[b * DMODEL + d];
  __syncthreads();
  for (int o = tid; o < NHEADS * DLAT; o += 256) {
    int h = o >> 9, l = o & 511;
    float accv = 0.f;
    const float* wr = Wkv_up + (size_t)l * (2 * DMODEL) + h * DHEAD;
    for (int d = 0; d < DHEAD; ++d) accv += qs[h * DHEAD + d] * wr[d];
    q_abs[((size_t)b * NHEADS + h) * DLAT + l] = (bf16_t)accv;
  }
  if (tid < NHEADS) {
    float accv = 0.f;
    for (int d = 0; d < DHEAD; ++d)
      accv += qs[tid * DHEAD + d] * bkv_up[tid * DHEAD + d];
    qb[b * NHEADS + tid] = accv;
  }
}

// ---------------------------------------------------------------------------
// 7) logits[b,h,k] = (q_abs[b,h,:] . K_down[b,idx[k],:] + qb) / sqrt(128)
//    WMMA: M=16 heads, N=16 selected tokens, K=512 latent (16 steps).
// ---------------------------------------------------------------------------
__global__ __launch_bounds__(256) void k_logits(const bf16_t* __restrict__ kv,
                                                const bf16_t* __restrict__ q_abs,
                                                const float* __restrict__ qb,
                                                const int* __restrict__ idx,
                                                float* __restrict__ logits) {
  const int kc = blockIdx.x;                      // 256-token chunk, 0..7
  const int b  = blockIdx.y;
  const int tid = threadIdx.x, wave = tid >> 5, lane = tid & 31;
  const int nloc = lane & 15;
  const int khA  = (lane < 16) ? 0 : 8;
  const int khB  = (lane < 16) ? 0 : 8;
  const bf16_t* qa = q_abs + (size_t)b * NHEADS * DLAT;
  const float scale = 0.08838834764831845f;       // 1/sqrt(128)
#pragma unroll
  for (int g = 0; g < 2; ++g) {
    const int tbase = kc * 256 + (wave * 2 + g) * 16;
    const int tok = idx[b * TOPK + tbase + nloc];
    const bf16_t* krow = kv + (size_t)(b * SEQ + tok) * KVN;  // K_down part
    v8f acc = vzero8();
#pragma unroll
    for (int ks = 0; ks < DLAT / 32; ++ks) {
      const bf16_t* ap = qa + (size_t)nloc * DLAT + ks * 32 + khA;
      v16bf afrag = cat16(*(const v8bf*)ap, *(const v8bf*)(ap + 16));
      const bf16_t* bp = krow + ks * 32 + khB;
      v16bf bfrag = cat16(*(const v8bf*)bp, *(const v8bf*)(bp + 16));
      acc = wmma_bf16f32(afrag, bfrag, acc);
    }
#pragma unroll
    for (int r = 0; r < 8; ++r) {
      int h = r + ((lane < 16) ? 0 : 8);
      logits[((size_t)b * NHEADS + h) * TOPK + tbase + nloc] =
          (acc[r] + qb[b * NHEADS + h]) * scale;
    }
  }
}

// ---------------------------------------------------------------------------
// 8) Softmax per (b,h) row; store unnormalized exp as bf16 + 1/sum.
// ---------------------------------------------------------------------------
__global__ __launch_bounds__(256) void k_softmax(const float* __restrict__ logits,
                                                 bf16_t* __restrict__ attn,
                                                 float* __restrict__ inv_s) {
  const int h = blockIdx.x, b = blockIdx.y, tid = threadIdx.x;
  const float* row = logits + ((size_t)b * NHEADS + h) * TOPK;
  __shared__ float red[256];
  float m = -1e30f;
  for (int i = tid; i < TOPK; i += 256) m = fmaxf(m, row[i]);
  red[tid] = m;
  __syncthreads();
  for (int s = 128; s > 0; s >>= 1) {
    if (tid < s) red[tid] = fmaxf(red[tid], red[tid + s]);
    __syncthreads();
  }
  m = red[0];
  __syncthreads();
  float sum = 0.f;
  for (int i = tid; i < TOPK; i += 256) {
    float e = __expf(row[i] - m);
    attn[((size_t)b * NHEADS + h) * TOPK + i] = (bf16_t)e;
    sum += e;
  }
  red[tid] = sum;
  __syncthreads();
  for (int s = 128; s > 0; s >>= 1) {
    if (tid < s) red[tid] += red[tid + s];
    __syncthreads();
  }
  if (tid == 0) inv_s[b * NHEADS + h] = 1.0f / red[0];
}

// ---------------------------------------------------------------------------
// 9) o_lat[b,h,l] = (1/sum) * sum_k exp[h,k] * V_down[b,idx[k],l]
//    Gather 32 V rows/step into LDS. Async-DMA path when the toolchain
//    exposes GLOBAL_LOAD_ASYNC_TO_LDS (per-lane gathered LDS fill, ASYNCcnt),
//    else manual transpose staging.
// ---------------------------------------------------------------------------
__global__ __launch_bounds__(256) void k_olat(const bf16_t* __restrict__ kv,
                                              const bf16_t* __restrict__ attn,
                                              const int* __restrict__ idx,
                                              const float* __restrict__ inv_s,
                                              float* __restrict__ o_lat) {
  const int lc = blockIdx.x;                      // 0..3 (latent chunk of 128)
  const int b  = blockIdx.y;
  const int tid = threadIdx.x, wave = tid >> 5, lane = tid & 31;
  const int nloc = lane & 15;
  const int khA  = (lane < 16) ? 0 : 8;
  const int khB  = (lane < 16) ? 0 : 8;
  v8f acc = vzero8();

#if HAVE_ASYNC_LDS
  __shared__ __align__(16) bf16_t Vg[32][128];    // [token_local][latent_local]
  for (int k0 = 0; k0 < TOPK; k0 += 32) {
#pragma unroll
    for (int i = 0; i < 2; ++i) {
      int c   = tid + i * 256;                    // 0..511 (32 rows x 16 chunks)
      int row = c >> 4;
      int ch  = (c & 15) << 3;
      int tok = idx[b * TOPK + k0 + row];
      const bf16_t* src =
          kv + (size_t)(b * SEQ + tok) * KVN + DLAT + lc * 128 + ch;
      __builtin_amdgcn_global_load_async_to_lds_b128(
          (as1_v4i_ptr)(void*)src,
          (as3_v4i_ptr)(void*)&Vg[row][ch], 0, 0);
    }
    wait_async0();
    __syncthreads();
    const bf16_t* ap = attn + ((size_t)b * NHEADS + nloc) * TOPK + k0 + khA;
    v16bf afrag = cat16(*(const v8bf*)ap, *(const v8bf*)(ap + 16));
    v16bf bfrag;
#pragma unroll
    for (int j = 0; j < 16; ++j) bfrag[j] = Vg[khB + j + (j & 8)][wave * 16 + nloc];
    acc = wmma_bf16f32(afrag, bfrag, acc);
    __syncthreads();
  }
#else
  __shared__ __align__(16) bf16_t Vt[128][32];    // [latent_local][token_local]
  const int tgl  = tid >> 3;                      // token_local 0..31
  const int loff = (tid & 7) << 4;                // latent offset 0..112
  for (int k0 = 0; k0 < TOPK; k0 += 32) {
    int tok = idx[b * TOPK + k0 + tgl];
    const bf16_t* vrow =
        kv + (size_t)(b * SEQ + tok) * KVN + DLAT + lc * 128 + loff;
    v8bf r0 = *(const v8bf*)vrow;
    v8bf r1 = *(const v8bf*)(vrow + 8);
#pragma unroll
    for (int j = 0; j < 8; ++j) Vt[loff + j][tgl] = r0[j];
#pragma unroll
    for (int j = 0; j < 8; ++j) Vt[loff + 8 + j][tgl] = r1[j];
    __syncthreads();
    const bf16_t* ap = attn + ((size_t)b * NHEADS + nloc) * TOPK + k0 + khA;
    v16bf afrag = cat16(*(const v8bf*)ap, *(const v8bf*)(ap + 16));
    const bf16_t* bp = &Vt[wave * 16 + nloc][khB];
    v16bf bfrag = cat16(*(const v8bf*)bp, *(const v8bf*)(bp + 16));
    acc = wmma_bf16f32(afrag, bfrag, acc);
    __syncthreads();
  }
#endif

#pragma unroll
  for (int r = 0; r < 8; ++r) {
    int h = r + ((lane < 16) ? 0 : 8);
    int l = lc * 128 + wave * 16 + nloc;
    o_lat[((size_t)b * NHEADS + h) * DLAT + l] =
        acc[r] * inv_s[b * NHEADS + h];
  }
}

// ---------------------------------------------------------------------------
// 10) o[b, h*128+d] = o_lat[b,h,:] . Wv_up[:, h*128+d] + bv_up
// ---------------------------------------------------------------------------
__global__ __launch_bounds__(256) void k_ov(const float* __restrict__ o_lat,
                                            const float* __restrict__ Wkv_up,
                                            const float* __restrict__ bkv_up,
                                            float* __restrict__ o) {
  const int b = blockIdx.y;
  const int j = blockIdx.x * 256 + threadIdx.x;   // 0..2047
  const int h = j >> 7;
  float accv = bkv_up[DMODEL + j];
  const float* olr = o_lat + ((size_t)b * NHEADS + h) * DLAT;
  for (int l = 0; l < DLAT; ++l)
    accv += olr[l] * Wkv_up[(size_t)l * (2 * DMODEL) + DMODEL + j];
  o[b * DMODEL + j] = accv;
}

// ---------------------------------------------------------------------------
// 11) out = o @ Wout + bout
// ---------------------------------------------------------------------------
__global__ __launch_bounds__(256) void k_out(const float* __restrict__ o,
                                             const float* __restrict__ Wout,
                                             const float* __restrict__ bout,
                                             float* __restrict__ out) {
  const int b = blockIdx.y;
  const int n = blockIdx.x * 256 + threadIdx.x;
  __shared__ float os[DMODEL];
  for (int d = threadIdx.x; d < DMODEL; d += 256) os[d] = o[b * DMODEL + d];
  __syncthreads();
  float accv = bout[n];
  for (int d = 0; d < DMODEL; ++d) accv += os[d] * Wout[(size_t)d * DMODEL + n];
  out[b * DMODEL + n] = accv;
}

// ---------------------------------------------------------------------------
// Launch
// ---------------------------------------------------------------------------
extern "C" void kernel_launch(void* const* d_in, const int* in_sizes, int n_in,
                              void* d_out, int out_size, void* d_ws,
                              size_t ws_size, hipStream_t stream) {
  const float* x        = (const float*)d_in[0];
  const float* Wq       = (const float*)d_in[1];
  const float* bq       = (const float*)d_in[2];
  const float* Wkv_down = (const float*)d_in[3];
  const float* bkv_down = (const float*)d_in[4];
  const float* Wq_down  = (const float*)d_in[5];
  const float* bq_down  = (const float*)d_in[6];
  const float* Wkv_up   = (const float*)d_in[7];
  const float* bkv_up   = (const float*)d_in[8];
  const float* Wout     = (const float*)d_in[9];
  const float* bout     = (const float*)d_in[10];
  float* out = (float*)d_out;
  (void)in_sizes; (void)n_in; (void)out_size; (void)ws_size;

  char* ws = (char*)d_ws;
  size_t off = 0;
  auto wsalloc = [&](size_t bytes) -> void* {
    off = (off + 255) & ~(size_t)255;
    void* p = ws + off;
    off += bytes;
    return p;
  };
  bf16_t* Wt     = (bf16_t*)wsalloc((size_t)KVN * DMODEL * sizeof(bf16_t));  // 4 MB
  bf16_t* kv     = (bf16_t*)wsalloc((size_t)MROWS * KVN * sizeof(bf16_t));   // 134 MB
  float*  scores = (float*) wsalloc((size_t)NBATCH * SEQ * 4);
  int*    idx    = (int*)   wsalloc((size_t)NBATCH * TOPK * 4);
  float*  q      = (float*) wsalloc((size_t)NBATCH * DMODEL * 4);
  float*  q_idx  = (float*) wsalloc((size_t)NBATCH * DLAT * 4);
  bf16_t* q_abs  = (bf16_t*)wsalloc((size_t)NBATCH * NHEADS * DLAT * 2);
  float*  qb     = (float*) wsalloc(NBATCH * NHEADS * 4);
  float*  logits = (float*) wsalloc((size_t)NBATCH * NHEADS * TOPK * 4);
  bf16_t* attn   = (bf16_t*)wsalloc((size_t)NBATCH * NHEADS * TOPK * 2);
  float*  inv_s  = (float*) wsalloc(NBATCH * NHEADS * 4);
  float*  o_lat  = (float*) wsalloc((size_t)NBATCH * NHEADS * DLAT * 4);
  float*  o      = (float*) wsalloc((size_t)NBATCH * DMODEL * 4);

  k_wprep  <<<dim3(KVN / 64),               256, 0, stream>>>(Wkv_down, Wt);
  k_kvdown <<<dim3(KVN / 128, MROWS / 128), 256, 0, stream>>>(x, Wt, bkv_down, kv);
  k_qidx   <<<dim3(NBATCH),                 256, 0, stream>>>(x, Wq_down, bq_down, q_idx);
  k_scores <<<dim3(SEQ / 256, NBATCH),      256, 0, stream>>>(kv, q_idx, scores);
  k_topk   <<<dim3(NBATCH),                1024, 0, stream>>>(scores, idx);
  k_qproj  <<<dim3(DMODEL / 256, NBATCH),   256, 0, stream>>>(x, Wq, bq, q);
  k_qabs   <<<dim3(NBATCH),                 256, 0, stream>>>(q, Wkv_up, bkv_up, q_abs, qb);
  k_logits <<<dim3(TOPK / 256, NBATCH),     256, 0, stream>>>(kv, q_abs, qb, idx, logits);
  k_softmax<<<dim3(NHEADS, NBATCH),         256, 0, stream>>>(logits, attn, inv_s);
  k_olat   <<<dim3(DLAT / 128, NBATCH),     256, 0, stream>>>(kv, attn, idx, inv_s, o_lat);
  k_ov     <<<dim3(DMODEL / 256, NBATCH),   256, 0, stream>>>(o_lat, Wkv_up, bkv_up, o);
  k_out    <<<dim3(DMODEL / 256, NBATCH),   256, 0, stream>>>(o, Wout, bout, out);
}
